// SwinTransformerBlock_82265803587941
// MI455X (gfx1250) — matrix operand order repaired and checked by reading, not compiled, not run on hardware
//
#include <hip/hip_runtime.h>
#include <hip/hip_bf16.h>
#include <math.h>

typedef __attribute__((ext_vector_type(16))) _Float16 v16h;
typedef __attribute__((ext_vector_type(8)))  float    v8f;
typedef __attribute__((ext_vector_type(4)))  int      v4i;

#define WMMA_F16(a, b, c) \
  __builtin_amdgcn_wmma_f32_16x16x32_f16(false, (a), false, (b), (short)0, (c), false, false)

// ---------------- static config (matches reference) ----------------
static constexpr int  CC       = 128;
static constexpr int  N_TOK    = 49;     // 7x7 tokens per window
static constexpr int  NPAD     = 64;     // padded to 4 WMMA M-tiles
static constexpr int  NHEAD    = 4;
static constexpr int  HD       = 32;
static constexpr int  NWIN_TOT = 2048;   // 32 batch * 64 windows
static constexpr int  MTOK     = 100352; // 32*56*56
static constexpr int  MLP_H    = 512;
static constexpr float SCALE_Q = 0.17677669529663687f; // 32^-0.5

// ---------------- CDNA5 async global->LDS staging ----------------
#if defined(__gfx1250__) && __has_builtin(__builtin_amdgcn_global_load_async_to_lds_b128)
#define USE_ASYNC_LDS 1
#else
#define USE_ASYNC_LDS 0
#endif

// copy 16 bytes (8 halves) global -> LDS
__device__ inline void cp16_g2l(_Float16* l, const _Float16* g) {
#if USE_ASYNC_LDS
  __builtin_amdgcn_global_load_async_to_lds_b128(
      (v4i*)(void*)g, (v4i*)(void*)l, 0, 0);
#else
  *(uint4*)(void*)l = *(const uint4*)(const void*)g;
#endif
}

__device__ inline void async_copy_fence() {
#if USE_ASYNC_LDS
#if __has_builtin(__builtin_amdgcn_s_wait_asynccnt)
  __builtin_amdgcn_s_wait_asynccnt(0);
#else
  asm volatile("s_wait_asynccnt 0" ::: "memory");
#endif
#endif
}

// ---------------- WMMA operand loaders (ISA 7.12.2 layouts) ----------------
// A tile: 16 rows x 32 K, f16 row-major source, leading dim ld.
// half h -> K = (h>>3)*16 + (lane>>4)*8 + (h&7), row = lane&15.  (2x16B per lane)
__device__ inline v16h load_a16(const _Float16* src, int ld, int lane) {
  int row = lane & 15;
  int kb  = (lane >> 4) << 3;
  v16h a;
#pragma unroll
  for (int h = 0; h < 16; ++h) {
    int k = ((h >> 3) << 4) + kb + (h & 7);
    a[h] = src[row * ld + k];
  }
  return a;
}
// B tile where logical B = src^T (src is [N x K] row-major) -> 2x16B per lane.
// half h -> K = (h>>3)*16 + (lane>>4)*8 + (h&7), col = lane&15.
__device__ inline v16h load_b_nxk(const _Float16* src, int ld, int lane) {
  int col = lane & 15;
  int kb  = (lane >> 4) << 3;
  v16h b;
#pragma unroll
  for (int h = 0; h < 16; ++h) b[h] = src[col * ld + ((h >> 3) << 4) + kb + (h & 7)];
  return b;
}

__device__ inline float gelu_exact(float v) {
  return 0.5f * v * (1.f + erff(v * 0.70710678118654752f));
}

// ---------------- K0: f32 -> f16 transpose-convert (weights to N-major) ----------------
// src: [K x N] f32 row-major  ->  dst: [N x K] f16 row-major
__global__ void __launch_bounds__(256) k_cvt_t(const float* __restrict__ src,
                                               _Float16* __restrict__ dst, int K, int N) {
  int i = blockIdx.x * blockDim.x + threadIdx.x;
  if (i < K * N) {
    int k = i / N, n = i % N;
    dst[n * K + k] = (_Float16)src[i];
  }
}

// ---------------- K1: LN1 + cyclic shift(-3,-3) + window partition ----------------
// output: xw [2048 win][64 tok][128] f16, rows 49..63 zero
__global__ void __launch_bounds__(256) k_ln1_part(const float* __restrict__ x,
                                                  const float* __restrict__ g,
                                                  const float* __restrict__ bb,
                                                  _Float16* __restrict__ xw) {
  int gid = blockIdx.x * blockDim.x + threadIdx.x;   // 2048*64 slots
  int win = gid >> 6, t = gid & 63;
  _Float16* dst = xw + (size_t)gid * CC;
  if (t >= N_TOK) {
    for (int c = 0; c < CC; ++c) dst[c] = (_Float16)0.f;
    return;
  }
  int bimg = win >> 6, w64 = win & 63;
  int wi = w64 >> 3, wj = w64 & 7;
  int th = t / 7, tw = t % 7;
  int hsrc = (wi * 7 + th + 3) % 56;
  int wsrc = (wj * 7 + tw + 3) % 56;
  const float* src = x + ((size_t)bimg * 3136 + hsrc * 56 + wsrc) * CC;
  float mu = 0.f, s2 = 0.f;
  for (int c = 0; c < CC; ++c) { float v = src[c]; mu += v; s2 += v * v; }
  mu *= (1.f / CC);
  s2 = s2 * (1.f / CC) - mu * mu;
  float r = rsqrtf(s2 + 1e-6f);
  for (int c = 0; c < CC; ++c)
    dst[c] = (_Float16)(((src[c] - mu) * r) * g[c] + bb[c]);
}

// ---------------- K2: QKV GEMM [64x128]@[128x384] per window (WMMA) ----------------
// q/k: [win][head][64 tok][32 d] f16 (Q pre-scaled);  v: [win][head][32 d][64 tok] f16
__global__ void __launch_bounds__(256) k_qkv(const _Float16* __restrict__ xw,
                                             const _Float16* __restrict__ wqT,
                                             const float* __restrict__ qkvb,
                                             _Float16* __restrict__ qb,
                                             _Float16* __restrict__ kb,
                                             _Float16* __restrict__ vbT) {
  __shared__ __align__(16) _Float16 As[NPAD * CC];   // 16 KB A panel
  int win = blockIdx.x;
  int tid = threadIdx.x, lane = tid & 31, wv = tid >> 5;  // 8 waves
  const _Float16* A0 = xw + (size_t)win * NPAD * CC;
  for (int i = tid * 8; i < NPAD * CC; i += 256 * 8) cp16_g2l(&As[i], A0 + i);
  async_copy_fence();
  __syncthreads();

  for (int mt = 0; mt < 4; ++mt) {
    v16h a[4];
#pragma unroll
    for (int j = 0; j < 4; ++j) a[j] = load_a16(&As[(mt * 16) * CC + j * 32], CC, lane);
    for (int nt = wv * 3; nt < wv * 3 + 3; ++nt) {        // 24 N-tiles total
      const _Float16* Bp = wqT + (size_t)(nt * 16) * CC;
      v8f acc = {};
#pragma unroll
      for (int j = 0; j < 4; ++j)
        acc = WMMA_F16(a[j], load_b_nxk(Bp + j * 32, CC, lane), acc);

      int row0 = mt * 16 + ((lane >> 4) << 3);
      int col  = nt * 16 + (lane & 15);
      int which = col >> 7, head = (col >> 5) & 3, d = col & 31;
      float bias = qkvb[col];
      if (which == 2) {  // V: store transposed (d-major) -> contiguous per lane
        size_t o = (((size_t)win * NHEAD + head) * HD + d) * NPAD + row0;
#pragma unroll
        for (int i = 0; i < 8; ++i) vbT[o + i] = (_Float16)(acc[i] + bias);
      } else {
        _Float16* dst = (which == 0) ? qb : kb;
        float sc = (which == 0) ? SCALE_Q : 1.f;
        size_t o = (((size_t)win * NHEAD + head) * NPAD + row0) * HD + d;
#pragma unroll
        for (int i = 0; i < 8; ++i)
          dst[o + (size_t)i * HD] = (_Float16)((acc[i] + bias) * sc);
      }
    }
  }
}

// ---------------- K3: fused attention per (window, head) ----------------
// S = QK^T (+rel-pos bias, +shift mask), double softmax (faithful), O = P@V
__global__ void __launch_bounds__(64) k_attn(const _Float16* __restrict__ qb,
                                             const _Float16* __restrict__ kb,
                                             const _Float16* __restrict__ vbT,
                                             const float* __restrict__ rpb,
                                             _Float16* __restrict__ ob) {
  __shared__ float S[NPAD][68];
  int win = blockIdx.x >> 2;
  int head = blockIdx.x & 3;
  int tid = threadIdx.x, lane = tid & 31, wv = tid >> 5;  // 2 waves of 32
  size_t baseQK = ((size_t)win * NHEAD + head) * NPAD * HD;
  size_t baseV  = ((size_t)win * NHEAD + head) * HD * NPAD;
  int widx = win & 63, wi = widx >> 3, wj = widx & 7;

  // ---- S = Q @ K^T, bias + mask fused at store ----
  for (int mt = wv * 2; mt < wv * 2 + 2; ++mt) {
    v16h a = load_a16(qb + baseQK + (mt * 16) * HD, HD, lane);
    for (int nt = 0; nt < 4; ++nt) {
      v16h b = load_b_nxk(kb + baseQK + (nt * 16) * HD, HD, lane);
      v8f acc = {};
      acc = WMMA_F16(a, b, acc);
      int r0 = mt * 16 + ((lane >> 4) << 3);
      int c  = nt * 16 + (lane & 15);
#pragma unroll
      for (int i = 0; i < 8; ++i) {
        int r = r0 + i;
        float v = acc[i];
        if (c >= N_TOK) {
          v = -1e30f;                         // key padding
        } else if (r < N_TOK) {
          int rh = r / 7, rw = r % 7, ch = c / 7, cw = c % 7;
          v += rpb[((rh - ch + 6) * 13 + (rw - cw + 6)) * NHEAD + head];
          int hr = wi * 7 + rh, wr = wj * 7 + rw;
          int hc = wi * 7 + ch, wc = wj * 7 + cw;
          int lr = ((hr < 49) ? 0 : (hr < 53) ? 1 : 2) * 3 + ((wr < 49) ? 0 : (wr < 53) ? 1 : 2);
          int lc = ((hc < 49) ? 0 : (hc < 53) ? 1 : 2) * 3 + ((wc < 49) ? 0 : (wc < 53) ? 1 : 2);
          if (lr != lc) v += -100.f;
        }
        S[r][c] = v;
      }
    }
  }
  __syncthreads();

  // ---- double softmax (reference applies softmax twice), one row per thread ----
  {
    int r = tid;  // 64 rows, 64 threads
    for (int pass = 0; pass < 2; ++pass) {
      float mx = -3e38f;
      for (int c = 0; c < NPAD; ++c) mx = fmaxf(mx, S[r][c]);
      float sum = 0.f;
      for (int c = 0; c < NPAD; ++c) sum += __expf(S[r][c] - mx);
      float inv = 1.f / sum;
      for (int c = 0; c < NPAD; ++c) {
        float p = __expf(S[r][c] - mx) * inv;
        // re-mask padded keys between the two softmaxes (ref has only 49 cols)
        S[r][c] = (pass == 0 && c >= N_TOK) ? -1e30f : p;
      }
    }
  }
  __syncthreads();

  // ---- O = P @ V  (V is d-major -> contiguous B loads) ----
  for (int mt = wv * 2; mt < wv * 2 + 2; ++mt) {
    for (int ntd = 0; ntd < 2; ++ntd) {
      v8f acc = {};
#pragma unroll
      for (int kt = 0; kt < 2; ++kt) {
        v16h a;
        int row = lane & 15, kbse = (lane >> 4) << 3;
#pragma unroll
        for (int h = 0; h < 16; ++h) {
          int k = ((h >> 3) << 4) + kbse + (h & 7) + kt * 32;
          a[h] = (_Float16)S[mt * 16 + row][k];
        }
        v16h b = load_b_nxk(vbT + baseV + (size_t)(ntd * 16) * NPAD + kt * 32, NPAD, lane);
        acc = WMMA_F16(a, b, acc);
      }
      int r0 = mt * 16 + ((lane >> 4) << 3);
      int d  = ntd * 16 + (lane & 15);
#pragma unroll
      for (int i = 0; i < 8; ++i)
        ob[((size_t)win * NPAD + r0 + i) * CC + head * HD + d] = (_Float16)acc[i];
    }
  }
}

// ---------------- K4: proj GEMM + LN + GELU + reverse shift + residual ----------------
__global__ void __launch_bounds__(128) k_proj(const _Float16* __restrict__ ob,
                                              const _Float16* __restrict__ wpT,
                                              const float* __restrict__ pb,
                                              const float* __restrict__ lng,
                                              const float* __restrict__ lnb,
                                              const float* __restrict__ x_in,
                                              float* __restrict__ x1) {
  __shared__ __align__(16) _Float16 As[NPAD * CC];   // 16 KB A panel
  __shared__ float Y[NPAD][132];
  int win = blockIdx.x;
  int tid = threadIdx.x, lane = tid & 31, wv = tid >> 5;  // 4 waves
  const _Float16* A0 = ob + (size_t)win * NPAD * CC;
  for (int i = tid * 8; i < NPAD * CC; i += 128 * 8) cp16_g2l(&As[i], A0 + i);
  async_copy_fence();
  __syncthreads();

  for (int mt = 0; mt < 4; ++mt) {
    v16h a[4];
#pragma unroll
    for (int j = 0; j < 4; ++j) a[j] = load_a16(&As[(mt * 16) * CC + j * 32], CC, lane);
    for (int nt = wv * 2; nt < wv * 2 + 2; ++nt) {
      const _Float16* Bp = wpT + (size_t)(nt * 16) * CC;
      v8f acc = {};
#pragma unroll
      for (int j = 0; j < 4; ++j)
        acc = WMMA_F16(a[j], load_b_nxk(Bp + j * 32, CC, lane), acc);
      int r0 = mt * 16 + ((lane >> 4) << 3);
      int col = nt * 16 + (lane & 15);
#pragma unroll
      for (int i = 0; i < 8; ++i) Y[r0 + i][col] = acc[i] + pb[col];
    }
  }
  __syncthreads();
  if (tid < N_TOK) {
    int t = tid;
    float mu = 0.f, s2 = 0.f;
    for (int c = 0; c < CC; ++c) { float v = Y[t][c]; mu += v; s2 += v * v; }
    mu *= (1.f / CC);
    s2 = s2 * (1.f / CC) - mu * mu;
    float r = rsqrtf(s2 + 1e-6f);
    int bimg = win >> 6, w64 = win & 63, wi = w64 >> 3, wj = w64 & 7;
    int th = t / 7, tw = t % 7;
    int hf = (wi * 7 + th + 3) % 56;     // reverse cyclic shift (+3,+3)
    int wf = (wj * 7 + tw + 3) % 56;
    size_t o = ((size_t)bimg * 3136 + hf * 56 + wf) * CC;
    for (int c = 0; c < CC; ++c) {
      float v = (Y[t][c] - mu) * r * lng[c] + lnb[c];
      x1[o + c] = x_in[o + c] + gelu_exact(v);
    }
  }
}

// ---------------- K5: LN2 -> f16 ----------------
__global__ void __launch_bounds__(256) k_ln2(const float* __restrict__ x1,
                                             const float* __restrict__ g,
                                             const float* __restrict__ bb,
                                             _Float16* __restrict__ m) {
  int t = blockIdx.x * blockDim.x + threadIdx.x;  // 100352 tokens
  const float* src = x1 + (size_t)t * CC;
  _Float16* dst = m + (size_t)t * CC;
  float mu = 0.f, s2 = 0.f;
  for (int c = 0; c < CC; ++c) { float v = src[c]; mu += v; s2 += v * v; }
  mu *= (1.f / CC);
  s2 = s2 * (1.f / CC) - mu * mu;
  float r = rsqrtf(s2 + 1e-6f);
  for (int c = 0; c < CC; ++c)
    dst[c] = (_Float16)(((src[c] - mu) * r) * g[c] + bb[c]);
}

// ---------------- K6: MLP1 GEMM [64x128]@[128x512] + GELU ----------------
__global__ void __launch_bounds__(128) k_mlp1(const _Float16* __restrict__ m,
                                              const _Float16* __restrict__ w1T,
                                              const float* __restrict__ b1,
                                              _Float16* __restrict__ h1) {
  __shared__ __align__(16) _Float16 As[NPAD * CC];   // 16 KB A panel (shared by 4 waves)
  int rowb = blockIdx.x * 64;
  int colb = blockIdx.y * 128;
  int tid = threadIdx.x, lane = tid & 31, wv = tid >> 5;  // 4 waves
  const _Float16* A0 = m + (size_t)rowb * CC;
  for (int i = tid * 8; i < NPAD * CC; i += 128 * 8) cp16_g2l(&As[i], A0 + i);
  async_copy_fence();
  __syncthreads();

  for (int mt = 0; mt < 4; ++mt) {
    v16h a[4];
#pragma unroll
    for (int j = 0; j < 4; ++j) a[j] = load_a16(&As[(mt * 16) * CC + j * 32], CC, lane);
    for (int nt = wv * 2; nt < wv * 2 + 2; ++nt) {
      int col0 = colb + nt * 16;
      const _Float16* Bp = w1T + (size_t)col0 * CC;
      v8f acc = {};
#pragma unroll
      for (int j = 0; j < 4; ++j)
        acc = WMMA_F16(a[j], load_b_nxk(Bp + j * 32, CC, lane), acc);
      int r0 = rowb + mt * 16 + ((lane >> 4) << 3);
      int col = col0 + (lane & 15);
      float bias = b1[col];
#pragma unroll
      for (int i = 0; i < 8; ++i) {
        float v = acc[i] + bias;
        h1[(size_t)(r0 + i) * MLP_H + col] = (_Float16)gelu_exact(v);
      }
    }
  }
}

// ---------------- K7: MLP2 GEMM [64x512]@[512x128] + residual -> out ----------------
__global__ void __launch_bounds__(128) k_mlp2(const _Float16* __restrict__ h1,
                                              const _Float16* __restrict__ w2T,
                                              const float* __restrict__ b2,
                                              const float* __restrict__ x1,
                                              float* __restrict__ out) {
  __shared__ __align__(16) _Float16 As[NPAD * MLP_H];  // 64 KB A panel (shared by 4 waves)
  int rowb = blockIdx.x * 64;
  int tid = threadIdx.x, lane = tid & 31, wv = tid >> 5;  // 4 waves
  int nt0 = wv * 2, nt1 = wv * 2 + 1;
  const _Float16* A0 = h1 + (size_t)rowb * MLP_H;
  for (int i = tid * 8; i < NPAD * MLP_H; i += 128 * 8) cp16_g2l(&As[i], A0 + i);
  async_copy_fence();
  __syncthreads();

  const _Float16* B0 = w2T + (size_t)(nt0 * 16) * MLP_H;
  const _Float16* B1 = w2T + (size_t)(nt1 * 16) * MLP_H;
  for (int mt = 0; mt < 4; ++mt) {
    v8f acc0 = {}, acc1 = {};
    for (int kc = 0; kc < MLP_H; kc += 128) {   // K chunked: 4 A tiles live
      __builtin_prefetch(B0 + kc + 128, 0, 1);
      v16h a[4];
#pragma unroll
      for (int j = 0; j < 4; ++j)
        a[j] = load_a16(&As[(mt * 16) * MLP_H + kc + j * 32], MLP_H, lane);
#pragma unroll
      for (int j = 0; j < 4; ++j) {
        int kk = kc + j * 32;
        acc0 = WMMA_F16(a[j], load_b_nxk(B0 + kk, MLP_H, lane), acc0);
        acc1 = WMMA_F16(a[j], load_b_nxk(B1 + kk, MLP_H, lane), acc1);
      }
    }
    int r0 = rowb + mt * 16 + ((lane >> 4) << 3);
    int col0 = nt0 * 16 + (lane & 15);
    int col1 = nt1 * 16 + (lane & 15);
    float bias0 = b2[col0], bias1 = b2[col1];
#pragma unroll
    for (int i = 0; i < 8; ++i) {
      size_t o0 = (size_t)(r0 + i) * CC + col0;
      size_t o1 = (size_t)(r0 + i) * CC + col1;
      out[o0] = x1[o0] + acc0[i] + bias0;
      out[o1] = x1[o1] + acc1[i] + bias1;
    }
  }
}

// ---------------- host orchestration ----------------
extern "C" void kernel_launch(void* const* d_in, const int* in_sizes, int n_in,
                              void* d_out, int out_size, void* d_ws, size_t ws_size,
                              hipStream_t stream) {
  const float* x       = (const float*)d_in[0];
  const float* qkv_w   = (const float*)d_in[1];
  const float* qkv_b   = (const float*)d_in[2];
  const float* proj_w  = (const float*)d_in[3];
  const float* proj_b  = (const float*)d_in[4];
  const float* pln_g   = (const float*)d_in[5];
  const float* pln_b   = (const float*)d_in[6];
  const float* rpb     = (const float*)d_in[7];
  const float* n1_g    = (const float*)d_in[8];
  const float* n1_b    = (const float*)d_in[9];
  const float* n2_g    = (const float*)d_in[10];
  const float* n2_b    = (const float*)d_in[11];
  const float* mlp_w1  = (const float*)d_in[12];
  const float* mlp_b1  = (const float*)d_in[13];
  const float* mlp_w2  = (const float*)d_in[14];
  const float* mlp_b2  = (const float*)d_in[15];
  float* out = (float*)d_out;

  char* ws = (char*)d_ws;
  size_t off = 0;
  auto carve = [&](size_t bytes) {
    size_t o = off;
    off += (bytes + 255) & ~(size_t)255;
    return o;
  };
  _Float16* wqT  = (_Float16*)(ws + carve((size_t)CC * 384 * 2));     // [384][128]
  _Float16* wpT  = (_Float16*)(ws + carve((size_t)CC * CC * 2));      // [128][128]
  _Float16* w1T  = (_Float16*)(ws + carve((size_t)CC * MLP_H * 2));   // [512][128]
  _Float16* w2T  = (_Float16*)(ws + carve((size_t)MLP_H * CC * 2));   // [128][512]
  _Float16* xw   = (_Float16*)(ws + carve((size_t)NWIN_TOT * NPAD * CC * 2));
  _Float16* qb   = (_Float16*)(ws + carve((size_t)NWIN_TOT * NHEAD * NPAD * HD * 2));
  _Float16* kb   = (_Float16*)(ws + carve((size_t)NWIN_TOT * NHEAD * NPAD * HD * 2));
  _Float16* vbT  = (_Float16*)(ws + carve((size_t)NWIN_TOT * NHEAD * HD * NPAD * 2));
  _Float16* ob   = (_Float16*)(ws + carve((size_t)NWIN_TOT * NPAD * CC * 2));
  float*    x1   = (float*)   (ws + carve((size_t)MTOK * CC * 4));
  _Float16* m16  = (_Float16*)(ws + carve((size_t)MTOK * CC * 2));
  _Float16* h1   = (_Float16*)(ws + carve((size_t)MTOK * MLP_H * 2));
  (void)ws_size; (void)n_in; (void)in_sizes; (void)out_size;

  // weight transpose-converts (f32 [K,N] -> f16 [N,K])
  k_cvt_t<<<(CC * 384 + 255) / 256, 256, 0, stream>>>(qkv_w, wqT, CC, 384);
  k_cvt_t<<<(CC * CC + 255) / 256, 256, 0, stream>>>(proj_w, wpT, CC, CC);
  k_cvt_t<<<(CC * MLP_H + 255) / 256, 256, 0, stream>>>(mlp_w1, w1T, CC, MLP_H);
  k_cvt_t<<<(MLP_H * CC + 255) / 256, 256, 0, stream>>>(mlp_w2, w2T, MLP_H, CC);

  // LN1 + shift + partition
  k_ln1_part<<<(NWIN_TOT * NPAD) / 256, 256, 0, stream>>>(x, n1_g, n1_b, xw);
  // QKV GEMM
  k_qkv<<<NWIN_TOT, 256, 0, stream>>>(xw, wqT, qkv_b, qb, kb, vbT);
  // fused attention per (window, head)
  k_attn<<<NWIN_TOT * NHEAD, 64, 0, stream>>>(qb, kb, vbT, rpb, ob);
  // proj + LN + GELU + reverse shift + residual
  k_proj<<<NWIN_TOT, 128, 0, stream>>>(ob, wpT, proj_b, pln_g, pln_b, x, x1);
  // LN2
  k_ln2<<<MTOK / 256, 256, 0, stream>>>(x1, n2_g, n2_b, m16);
  // MLP
  k_mlp1<<<dim3(MTOK / 64, MLP_H / 128), 128, 0, stream>>>(m16, w1T, mlp_b1, h1);
  k_mlp2<<<MTOK / 64, 128, 0, stream>>>(h1, w2T, mlp_b2, x1, out);
}
